// SparseAttention_16922171146856
// MI455X (gfx1250) — matrix operand order repaired and checked by reading, not compiled, not run on hardware
//
#include <hip/hip_runtime.h>
#include <math.h>

typedef __attribute__((ext_vector_type(16))) __bf16        v16bf;
typedef __attribute__((ext_vector_type(8)))  float         v8f;
typedef __attribute__((ext_vector_type(4)))  unsigned int  v4u;

// Problem constants
#define NPIX      36864          // 48*48*16 pixels
#define QKV_COLS  640            // Q:0-63, K:64-127, V:128-639

// Workspace layout (in floats)
#define OFF_QKV   0
#define SZ_QKV    (NPIX * QKV_COLS)          // 23,592,960
#define OFF_SH    (OFF_QKV + SZ_QKV)
#define SZ_SH     (768 * 48 * 48)            // 1,769,472
#define OFF_SW    (OFF_SH + SZ_SH)
#define OFF_ST    (OFF_SW + SZ_SH)
#define SZ_ST     (2304 * 16 * 16)           // 589,824
#define OFF_OUTW  (OFF_ST + SZ_ST)
#define SZ_OUTW   (NPIX * 512)               // 18,874,368
// total 46,596,096 floats = ~186.4 MB of d_ws

// ---------------- bf16 / WMMA helpers ----------------

__device__ __forceinline__ unsigned short f2bf(float f) {
  unsigned int x = __float_as_uint(f);
  unsigned int r = (x + 0x7fffu + ((x >> 16) & 1u)) >> 16;  // RNE
  return (unsigned short)r;
}

union Frag16 { v16bf v; unsigned short u[16]; v4u x4[2]; };

// A-matrix (16x32 bf16): lane m=L&15, half=L>>4.
// element e: K = k0 + half*8 + e (e<8) ; K = k0 + 16 + half*8 + (e-8) (e>=8)
// stride % 8 == 0 and k0 % 8 == 0  ->  both 8-ushort runs are 16B aligned.
__device__ __forceinline__ v16bf load_a_frag(const unsigned short* S, int stride,
                                             int row0, int k0, int lane) {
  int mm = lane & 15, half = lane >> 4;
  const v4u* p = (const v4u*)(S + (row0 + mm) * stride + k0 + half * 8);
  Frag16 f;
  f.x4[0] = p[0];   // K = half*8 + 0..7
  f.x4[1] = p[2];   // K = 16 + half*8 + 0..7   (+16 ushorts = +2 v4u)
  return f.v;
}

// B-matrix (32x16 bf16) staged TRANSPOSED as [n][k]: lane n=L&15, half=L>>4.
// element e: K = k0 + half*8 + e (e<8) ; K = k0 + 16 + half*8 + (e-8) (e>=8)
__device__ __forceinline__ v16bf load_b_frag(const unsigned short* St, int stride,
                                             int k0, int col0, int lane) {
  int nn = lane & 15, half = lane >> 4;
  const v4u* p = (const v4u*)(St + (col0 + nn) * stride + k0 + half * 8);
  Frag16 f;
  f.x4[0] = p[0];
  f.x4[1] = p[2];
  return f.v;
}

__device__ __forceinline__ v8f wmma_bf16(v16bf a, v16bf b, v8f c) {
  return __builtin_amdgcn_wmma_f32_16x16x32_bf16(false, a, false, b, (short)0, c,
                                                 false, false);
}

__device__ __forceinline__ float wred_max(float v) {
#pragma unroll
  for (int off = 16; off > 0; off >>= 1) v = fmaxf(v, __shfl_xor(v, off, 32));
  return v;
}
__device__ __forceinline__ float wred_sum(float v) {
#pragma unroll
  for (int off = 16; off > 0; off >>= 1) v += __shfl_xor(v, off, 32);
  return v;
}

// ---------------- Kernel 1: fused QKV projection GEMM ----------------
// out[p][j] = sum_c x[p][c] * W[j][c] + b[j]; 64-pixel x 64-outchan tiles.
__global__ __launch_bounds__(256) void k_proj(
    const float* __restrict__ x, const float* __restrict__ Wq,
    const float* __restrict__ bq, const float* __restrict__ Wk,
    const float* __restrict__ bk, const float* __restrict__ Wv,
    const float* __restrict__ bv, float* __restrict__ qkv) {
  __shared__ __attribute__((aligned(16))) unsigned short As[64 * 40];  // [pix][k]
  __shared__ __attribute__((aligned(16))) unsigned short Bt[64 * 40];  // [j][k] (natural W layout)
  int tid = threadIdx.x, lane = tid & 31, wv = tid >> 5;
  int m0 = blockIdx.x * 64;               // 2304 % 64 == 0 -> never crosses t
  int t = m0 / 2304, s0 = m0 % 2304;
  int by = blockIdx.y, j0 = by * 64;
  const float* Wsel; const float* bsel; int jr;
  if (by == 0)      { Wsel = Wq; bsel = bq; jr = 0; }
  else if (by == 1) { Wsel = Wk; bsel = bk; jr = 0; }
  else              { Wsel = Wv; bsel = bv; jr = j0 - 128; }

  int row0 = (wv & 3) * 16, col0 = (wv >> 2) * 32;  // 8 waves: 4x2 of (16x32)
  v8f acc0 = {}; v8f acc1 = {};
  int mA = tid & 63, kq = tid >> 6;  // A loader: lanes over pixels (coalesced)
  int kB = tid & 31, jq = tid >> 5;  // B loader: lanes over k (coalesced)
  for (int k0 = 0; k0 < 512; k0 += 32) {
#pragma unroll
    for (int q = 0; q < 8; ++q) {
      int k = kq * 8 + q;
      As[mA * 40 + k] = f2bf(x[(size_t)(t * 512 + k0 + k) * 2304 + s0 + mA]);
    }
#pragma unroll
    for (int q = 0; q < 8; ++q) {
      int j = jq * 8 + q;
      Bt[j * 40 + kB] = f2bf(Wsel[(size_t)(jr + j) * 512 + k0 + kB]);
    }
    __syncthreads();
    v16bf a  = load_a_frag(As, 40, row0, 0, lane);
    v16bf b0 = load_b_frag(Bt, 40, 0, col0, lane);
    v16bf b1 = load_b_frag(Bt, 40, 0, col0 + 16, lane);
    acc0 = wmma_bf16(a, b0, acc0);
    acc1 = wmma_bf16(a, b1, acc1);
    __syncthreads();
  }
  int nn = lane & 15, half = lane >> 4;
  float bias0 = bsel[jr + col0 + nn];
  float bias1 = bsel[jr + col0 + 16 + nn];
#pragma unroll
  for (int r = 0; r < 8; ++r) {
    int p = m0 + row0 + r + 8 * half;  // D layout: M = r + 8*half
    qkv[(size_t)p * 640 + j0 + col0 + nn]      = acc0[r] + bias0;
    qkv[(size_t)p * 640 + j0 + col0 + 16 + nn] = acc1[r] + bias1;
  }
}

// ---------------- Kernel 2a: H/W axis scores (48x48 Gram, K=64) ----------------
// axis0(H): g=w*16+t, p(r)=t*2304+r*48+w ; axis1(W): g=h*16+t, p(r)=t*2304+h*48+r
__global__ __launch_bounds__(288) void k_score48(const float* __restrict__ qkv,
                                                 float* __restrict__ sc, int axis) {
  __shared__ __attribute__((aligned(16))) unsigned short Qs[48 * 72];  // A: [query][c]
  __shared__ __attribute__((aligned(16))) unsigned short Kt[48 * 72];  // B^T: [key][c]
  int tid = threadIdx.x, lane = tid & 31, wv = tid >> 5;
  int g = blockIdx.x;
  int a1 = g >> 4, t = g & 15;
  for (int idx = tid; idx < 3072; idx += 288) {
    int r = idx >> 6, c = idx & 63;
    int p = (axis == 0) ? (t * 2304 + r * 48 + a1) : (t * 2304 + a1 * 48 + r);
    Qs[r * 72 + c] = f2bf(qkv[(size_t)p * 640 + c]);
    Kt[r * 72 + c] = f2bf(qkv[(size_t)p * 640 + 64 + c]);
  }
  __syncthreads();
  int mi = wv / 3, ni = wv % 3;  // 9 waves -> 3x3 tiles of 16x16
  v8f acc = {};
  for (int ks = 0; ks < 64; ks += 32) {
    v16bf a = load_a_frag(Qs, 72, mi * 16, ks, lane);
    v16bf b = load_b_frag(Kt, 72, ks, ni * 16, lane);
    acc = wmma_bf16(a, b, acc);
  }
  int nn = lane & 15, half = lane >> 4;
#pragma unroll
  for (int r = 0; r < 8; ++r)
    sc[(size_t)g * 2304 + (mi * 16 + r + 8 * half) * 48 + ni * 16 + nn] = acc[r];
}

// ---------------- Kernel 2b: T axis scores (16x16 Gram per (h,w)) ----------------
__global__ __launch_bounds__(256) void k_scoreT(const float* __restrict__ qkv,
                                                float* __restrict__ sT) {
  __shared__ __attribute__((aligned(16))) unsigned short Qs[8][16 * 72];
  __shared__ __attribute__((aligned(16))) unsigned short Kt[8][16 * 72];
  int tid = threadIdx.x, lane = tid & 31, wv = tid >> 5;
  int g = blockIdx.x * 8 + wv;  // 2304 groups, 8 per block
  int h = g / 48, w = g % 48;
  for (int idx = lane; idx < 1024; idx += 32) {
    int r = idx >> 6, c = idx & 63;
    int p = r * 2304 + h * 48 + w;
    Qs[wv][r * 72 + c] = f2bf(qkv[(size_t)p * 640 + c]);
    Kt[wv][r * 72 + c] = f2bf(qkv[(size_t)p * 640 + 64 + c]);
  }
  __syncthreads();
  v8f acc = {};
  for (int ks = 0; ks < 64; ks += 32) {
    v16bf a = load_a_frag(Qs[wv], 72, 0, ks, lane);
    v16bf b = load_b_frag(Kt[wv], 72, ks, 0, lane);
    acc = wmma_bf16(a, b, acc);
  }
  int nn = lane & 15, half = lane >> 4;
#pragma unroll
  for (int r = 0; r < 8; ++r)
    sT[(size_t)g * 256 + (r + 8 * half) * 16 + nn] = acc[r];
}

// ---------------- Kernel 3: softmax over 112-wide concat row, in place ----------------
__global__ __launch_bounds__(256) void k_softmax(float* __restrict__ sH,
                                                 float* __restrict__ sW,
                                                 float* __restrict__ sT) {
  int tid = threadIdx.x, lane = tid & 31, wv = tid >> 5;
  int P = blockIdx.x * 8 + wv;  // P = (h*48+w)*16+t
  int h = P / 768, rem = P % 768, w = rem / 16, t = rem % 16;
  size_t bH = ((size_t)(w * 16 + t) * 48 + h) * 48;
  size_t bW = ((size_t)(h * 16 + t) * 48 + w) * 48;
  size_t bT = ((size_t)(h * 48 + w) * 16 + t) * 16;
  float xv[4];
  float lmax = -3.4e38f;
#pragma unroll
  for (int q = 0; q < 4; ++q) {
    int j = lane + q * 32;
    float v = -3.4e38f;
    if (j < 48) v = sH[bH + j];
    else if (j < 96) v = sW[bW + j - 48];
    else if (j < 112) v = sT[bT + j - 96];
    xv[q] = v;
    lmax = fmaxf(lmax, v);
  }
  float vmax = wred_max(lmax);
  float lsum = 0.f;
#pragma unroll
  for (int q = 0; q < 4; ++q) {
    int j = lane + q * 32;
    float e = (j < 112) ? __expf(xv[q] - vmax) : 0.f;
    xv[q] = e; lsum += e;
  }
  float inv = 1.f / wred_sum(lsum);
#pragma unroll
  for (int q = 0; q < 4; ++q) {
    int j = lane + q * 32;
    float e = xv[q] * inv;
    if (j < 48) sH[bH + j] = e;
    else if (j < 96) sW[bW + j - 48] = e;
    else if (j < 112) sT[bT + j - 96] = e;
  }
}

// ---------------- Kernel 4: out = att x V per group (WMMA), 3 axis passes ----------------
// axis1 applies the reference's raw-reinterpret scramble of attW during the LDS gather.
__global__ __launch_bounds__(256) void k_out(const float* __restrict__ qkv,
                                             const float* __restrict__ att,
                                             float* __restrict__ outw, int axis) {
  __shared__ __attribute__((aligned(16))) unsigned short attS[48 * 72];  // A: [row][K]
  __shared__ __attribute__((aligned(16))) unsigned short Vt[128 * 72];   // B^T: [col][K]
  int tid = threadIdx.x, lane = tid & 31, wv = tid >> 5;
  int g = blockIdx.x;
  int L  = (axis == 2) ? 16 : 48;
  int KP = (axis == 2) ? 32 : 64;  // K padded to WMMA multiple
  int str = KP + 8;                // % 8 == 0 -> aligned b128 fragment loads
  int cb = blockIdx.y * 128;

  for (int idx = tid; idx < L * KP; idx += 256) {
    int rr = idx / KP, kk = idx % KP;
    float v = 0.f;
    if (kk < L) {
      if (axis == 0) {
        v = att[(size_t)g * 2304 + rr * 48 + kk];
      } else if (axis == 1) {  // scrambled attW gather (faithful reinterpret)
        int h = g >> 4, t = g & 15;
        v = att[(size_t)(h * 16 + (kk & 15)) * 2304 +
                (3 * (rr & 15) + (kk >> 4)) * 48 + (3 * t + (rr >> 4))];
      } else {
        v = att[(size_t)g * 256 + rr * 16 + kk];
      }
    }
    attS[rr * str + kk] = f2bf(v);
  }
  for (int idx = tid; idx < KP * 128; idx += 256) {
    int kk = idx / 128, cc = idx % 128;   // lanes over cc -> coalesced global read
    float v = 0.f;
    if (kk < L) {
      int p;
      if (axis == 0)      { int w = g >> 4, t = g & 15; p = t * 2304 + kk * 48 + w; }
      else if (axis == 1) { int h = g >> 4, t = g & 15; p = t * 2304 + h * 48 + kk; }
      else                { int h = g / 48, w = g % 48; p = kk * 2304 + h * 48 + w; }
      v = qkv[(size_t)p * 640 + 128 + cb + cc];
    }
    Vt[cc * str + kk] = f2bf(v);          // transposed stage: [col][K]
  }
  __syncthreads();

  int nMI = L / 16;
  int nn = lane & 15, half = lane >> 4;
  for (int mi = 0; mi < nMI; ++mi) {
    v8f acc = {};
    for (int ks = 0; ks < KP; ks += 32) {
      v16bf a = load_a_frag(attS, str, mi * 16, ks, lane);
      v16bf b = load_b_frag(Vt, str, ks, wv * 16, lane);
      acc = wmma_bf16(a, b, acc);
    }
#pragma unroll
    for (int r = 0; r < 8; ++r) {
      int mrow = mi * 16 + r + 8 * half;
      int p;
      if (axis == 0)      { int w = g >> 4, t = g & 15; p = t * 2304 + mrow * 48 + w; }
      else if (axis == 1) { int h = g >> 4, t = g & 15; p = t * 2304 + h * 48 + mrow; }
      else                { int h = g / 48, w = g % 48; p = mrow * 2304 + h * 48 + w; }
      size_t addr = (size_t)p * 512 + cb + wv * 16 + nn;
      if (axis == 0) outw[addr] = acc[r];       // first pass: plain store
      else           outw[addr] += acc[r];      // unique owner -> safe RMW
    }
  }
}

// ---------------- Kernel 5: objaff = max over query of mask*att ----------------
__global__ __launch_bounds__(256) void k_obj(const int* __restrict__ mcls,
                                             const float* __restrict__ sH,
                                             const float* __restrict__ sW,
                                             const float* __restrict__ sT,
                                             float* __restrict__ obj) {
  int idx = blockIdx.x * 256 + threadIdx.x;  // ((t*8+cls)*48+A)*48+B
  int B = idx % 48;
  int A = (idx / 48) % 48;
  int cls = (idx / 2304) % 8;
  int t = idx / 18432;
  float best = 0.f;  // all candidates are >= 0
  {  // H: mask from transpose(mo,(0,3,4,1,2)) raw reshape
    int g1 = B * 16 + t;
    int fb = g1 * 384 + cls * 48;
    for (int h = 0; h < 48; ++h) {
      int f = fb + h;
      int c = f & 7, tt = (f >> 3) & 15, ww = (f >> 7) % 48, hh = f / 6144;
      if (mcls[(tt * 48 + hh) * 48 + ww] == c)
        best = fmaxf(best, sH[(size_t)g1 * 2304 + h * 48 + A]);
    }
  }
  {  // W: mask from transpose(mo,(0,2,4,1,3)); att scrambled as in k_out axis1
    int g2 = A * 16 + t;
    int fb = g2 * 384 + cls * 48;
    size_t gbase = (size_t)(A * 16 + (B & 15)) * 2304;
    for (int r = 0; r < 48; ++r) {
      int f = fb + r;
      int hh = f % 48, tt = (f / 48) & 15, ww = (f / 768) % 48, c = f / 36864;
      if (mcls[(tt * 48 + hh) * 48 + ww] == c)
        best = fmaxf(best, sW[gbase + (3 * (r & 15) + (B >> 4)) * 48 +
                              (3 * t + (r >> 4))]);
    }
  }
  {  // T: mask from transpose(mo,(0,2,3,1,4))
    int g3 = A * 48 + B;
    int fb = g3 * 128 + cls * 16;
    for (int tq = 0; tq < 16; ++tq) {
      int f = fb + tq;
      int ww = f % 48, tt = (f / 48) & 15, hh = (f / 768) % 48, c = f / 36864;
      if (mcls[(tt * 48 + hh) * 48 + ww] == c)
        best = fmaxf(best, sT[(size_t)g3 * 256 + tq * 16 + t]);
    }
  }
  obj[idx] = best;
}

// ---------------- Kernel 6: finalize: (pix,c) -> (t,c,h,w), apply gamma ----------------
__global__ __launch_bounds__(256) void k_finalize(const float* __restrict__ outw,
                                                  const float* __restrict__ gamma,
                                                  float* __restrict__ out) {
  __shared__ float tile[64][65];
  int tid = threadIdx.x;
  int p0 = blockIdx.x * 64, c0 = blockIdx.y * 64;
  {
    int cl = tid & 63, pq = tid >> 6;
#pragma unroll
    for (int q = 0; q < 16; ++q) {
      int pl = pq * 16 + q;
      tile[pl][cl] = outw[(size_t)(p0 + pl) * 512 + c0 + cl];  // coalesced read
    }
  }
  __syncthreads();
  float gm = gamma[0];
  int pl = tid & 63, cq = tid >> 6;
  int p = p0 + pl;
  int t = p / 2304, hw = p % 2304;
#pragma unroll
  for (int q = 0; q < 16; ++q) {
    int c = cq * 16 + q;
    out[(size_t)(t * 512 + c0 + c) * 2304 + hw] = gm * tile[pl][c];  // coalesced write
  }
}

// ---------------- launcher ----------------
extern "C" void kernel_launch(void* const* d_in, const int* in_sizes, int n_in,
                              void* d_out, int out_size, void* d_ws, size_t ws_size,
                              hipStream_t stream) {
  (void)in_sizes; (void)n_in; (void)out_size; (void)ws_size;
  const float* x    = (const float*)d_in[0];
  const int*   mcls = (const int*)d_in[1];
  // d_in[2] = num_cls (constant 8, unused)
  const float* Wq = (const float*)d_in[3];
  const float* bq = (const float*)d_in[4];
  const float* Wk = (const float*)d_in[5];
  const float* bk = (const float*)d_in[6];
  const float* Wv = (const float*)d_in[7];
  const float* bv = (const float*)d_in[8];
  const float* gamma = (const float*)d_in[9];

  float* ws   = (float*)d_ws;
  float* qkv  = ws + OFF_QKV;
  float* sH   = ws + OFF_SH;
  float* sW   = ws + OFF_SW;
  float* sT   = ws + OFF_ST;
  float* outw = ws + OFF_OUTW;
  float* out_main = (float*)d_out;
  float* out_obj  = out_main + (size_t)16 * 512 * 48 * 48;  // 18,874,368

  k_proj<<<dim3(576, 10), 256, 0, stream>>>(x, Wq, bq, Wk, bk, Wv, bv, qkv);
  k_score48<<<dim3(768), 288, 0, stream>>>(qkv, sH, 0);
  k_score48<<<dim3(768), 288, 0, stream>>>(qkv, sW, 1);
  k_scoreT<<<dim3(288), 256, 0, stream>>>(qkv, sT);
  k_softmax<<<dim3(4608), 256, 0, stream>>>(sH, sW, sT);
  k_out<<<dim3(768, 4), 256, 0, stream>>>(qkv, sH, outw, 0);
  k_out<<<dim3(768, 4), 256, 0, stream>>>(qkv, sW, outw, 1);
  k_out<<<dim3(2304, 4), 256, 0, stream>>>(qkv, sT, outw, 2);
  k_obj<<<dim3(1152), 256, 0, stream>>>(mcls, sH, sW, sT, out_obj);
  k_finalize<<<dim3(576, 8), 256, 0, stream>>>(outw, gamma, out_main);
}
